// WaveGlow_39273180955482
// MI455X (gfx1250) — compile-verified, matrix operand order checked by reading, not run on hardware
//
#include <hip/hip_runtime.h>
#include <math.h>

typedef __attribute__((ext_vector_type(16))) _Float16 v16h;
typedef __attribute__((ext_vector_type(8)))  _Float16 v8h;
typedef __attribute__((ext_vector_type(8)))  float    v8f;

#define NB   2
#define TG   4096
#define NCH  256
#define COND 640
#define NMEL 80
#define KZ   1408      // 3*256 (dilated taps) + 640 (cond)

#define BM 128
#define BN 128
#define BK 32
#define LDA 40         // halves; 80 bytes -> multiple of 16B for b128 LDS ops
#define LDB 40
#define NKZ (KZ / BK)  // 44 K-steps for the fused z GEMM
#define NKR (NCH / BK) // 8 K-steps for the res-skip GEMM

// ---------------------------------------------------------------------------
// CDNA5 async global->LDS copy (ASYNCcnt) + drain
// ---------------------------------------------------------------------------
__device__ __forceinline__ void async_ld_b128(unsigned lds_off,
                                              unsigned long long gaddr) {
  asm volatile("global_load_async_to_lds_b128 %0, %1, off"
               :: "v"(lds_off), "v"(gaddr) : "memory");
}
__device__ __forceinline__ void wait_async0() {
  asm volatile("s_wait_asynccnt 0x0" ::: "memory");
}

// ---------------------------------------------------------------------------
// Upsample (transposed conv k=1024 stride=256) directly into cond layout
// s[b, m*8+g, t] = sum_i sum_j spect[b,i,j] * w[i, m, tt-256j], tt = t*8+g
// ---------------------------------------------------------------------------
__global__ void upsample_cond_kernel(const float* __restrict__ spect,
                                     const float* __restrict__ wups,
                                     float* __restrict__ s) {
  int idx = blockIdx.x * blockDim.x + threadIdx.x;
  if (idx >= NB * COND * TG) return;
  int t = idx & (TG - 1);
  int c = (idx >> 12) % COND;
  int b = idx / (COND * TG);
  int m = c >> 3, g = c & 7;
  int tt = t * 8 + g;
  int jlo = (tt - 768) >> 8; if (jlo < 0) jlo = 0;   // ceil((tt-1023)/256)
  int jhi = tt >> 8;        if (jhi > 127) jhi = 127;
  float acc = 0.f;
  for (int i = 0; i < NMEL; ++i) {
    const float* xp = spect + (b * NMEL + i) * 128;
    const float* wp = wups + (i * NMEL + m) * 1024;
    for (int j = jlo; j <= jhi; ++j)
      acc += xp[j] * wp[tt - (j << 8)];
  }
  s[idx] = acc;
}

__global__ void reshape_audio_kernel(const float* __restrict__ audio,
                                     float* __restrict__ a) {
  int idx = blockIdx.x * blockDim.x + threadIdx.x;
  if (idx >= NB * 8 * TG) return;
  int t = idx & (TG - 1);
  int g = (idx >> 12) & 7;
  int b = idx >> 15;
  a[idx] = audio[b * 32768 + t * 8 + g];
}

// ---------------------------------------------------------------------------
// Invertible 1x1 conv: aout[b,o,t] = sum_c W[o,c] * ain[b,ch0+c,t]
// a buffers are (NB, 8, TG) with fixed batch stride 8*TG
// ---------------------------------------------------------------------------
__global__ void invconv_kernel(const float* __restrict__ W,
                               const float* __restrict__ ain,
                               float* __restrict__ aout, int nr, int ch0) {
  int idx = blockIdx.x * blockDim.x + threadIdx.x;
  if (idx >= NB * TG) return;
  int t = idx & (TG - 1);
  int b = idx >> 12;
  float v[8];
  for (int c = 0; c < nr; ++c) v[c] = ain[(b * 8 + ch0 + c) * TG + t];
  for (int o = 0; o < nr; ++o) {
    float acc = 0.f;
    for (int c = 0; c < nr; ++c) acc += W[o * nr + c] * v[c];
    aout[(b * 8 + o) * TG + t] = acc;
  }
}

// log|det W| * B * Tg  (tiny LU with partial pivoting, single thread)
__global__ void logdet_kernel(const float* __restrict__ W, float* out, int nr) {
  if (threadIdx.x != 0 || blockIdx.x != 0) return;
  double m[64];
  for (int i = 0; i < nr * nr; ++i) m[i] = (double)W[i];
  double ld = 0.0;
  for (int k = 0; k < nr; ++k) {
    int p = k; double best = fabs(m[k * nr + k]);
    for (int r = k + 1; r < nr; ++r) {
      double v = fabs(m[r * nr + k]);
      if (v > best) { best = v; p = r; }
    }
    if (p != k)
      for (int c = 0; c < nr; ++c) {
        double tmp = m[k * nr + c]; m[k * nr + c] = m[p * nr + c]; m[p * nr + c] = tmp;
      }
    double piv = m[k * nr + k];
    ld += log(fabs(piv));
    for (int r = k + 1; r < nr; ++r) {
      double f = m[r * nr + k] / piv;
      for (int c = k; c < nr; ++c) m[r * nr + c] -= f * m[k * nr + c];
    }
  }
  *out = (float)((double)(NB * TG) * ld);
}

// start conv: x[b,co,t] = sum_h w[co*nh+h] * a[b,h,t]   (K <= 4)
__global__ void start_conv_kernel(const float* __restrict__ w,
                                  const float* __restrict__ a,
                                  float* __restrict__ x, int nh) {
  int idx = blockIdx.x * blockDim.x + threadIdx.x;
  if (idx >= NB * NCH * TG) return;
  int t = idx & (TG - 1);
  int co = (idx >> 12) & 255;
  int b = idx >> 20;
  float acc = 0.f;
  for (int h = 0; h < nh; ++h) acc += w[co * nh + h] * a[(b * 8 + h) * TG + t];
  x[idx] = acc;
}

// pack fused weight A[co, k] f16: k<768 -> in_w[co, k%256, k/256]; else cond_w
__global__ void pack_wz_kernel(const float* __restrict__ in_w,
                               const float* __restrict__ cond_w,
                               _Float16* __restrict__ A) {
  int idx = blockIdx.x * blockDim.x + threadIdx.x;
  if (idx >= 512 * KZ) return;
  int k = idx % KZ;
  int co = idx / KZ;
  float v;
  if (k < 768) {
    int kk = k >> 8, ci = k & 255;
    v = in_w[(co * NCH + ci) * 3 + kk];
  } else {
    v = cond_w[co * COND + (k - 768)];
  }
  A[idx] = (_Float16)v;
}

__global__ void pack_f16_kernel(const float* __restrict__ w,
                                _Float16* __restrict__ o, int n) {
  int i = blockIdx.x * blockDim.x + threadIdx.x;
  if (i < n) o[i] = (_Float16)w[i];
}

__global__ void zero_kernel(float* __restrict__ p, int n) {
  int i = blockIdx.x * blockDim.x + threadIdx.x;
  if (i < n) p[i] = 0.f;
}

// ---------------------------------------------------------------------------
// WMMA fragment helper: 16-bit A/B operand, lane group kb pattern per ISA:
// lanes 0-15 read K [0..8)+[16..24), lanes 16-31 read K [8..16)+[24..32)
// ---------------------------------------------------------------------------
__device__ __forceinline__ v16h load_frag(const _Float16* base, int ld,
                                          int row, int kb) {
  const v8h lo = *(const v8h*)(base + row * ld + kb);
  const v8h hi = *(const v8h*)(base + row * ld + kb + 16);
  v16h f;
#pragma unroll
  for (int q = 0; q < 8; ++q) { f[q] = lo[q]; f[q + 8] = hi[q]; }
  return f;
}

// ---------------------------------------------------------------------------
// Fused z = in_conv(x, dilated) + cond_conv(s): M=512, K=1408, N=8192
// Double-buffered LDS; weight tile via async global->LDS (ASYNCcnt),
// activation tile gathered+converted by VALU (packed to two b128 LDS stores)
// into the other buffer while WMMA consumes the current one.
// ---------------------------------------------------------------------------
__global__ __launch_bounds__(256)
void wn_z_gemm_kernel(const _Float16* __restrict__ A,  // [512, KZ] f16
                      const float* __restrict__ x,     // (NB,256,TG)
                      const float* __restrict__ s,     // (NB,640,TG)
                      float* __restrict__ z,           // (NB,512,TG)
                      int dil) {
  __shared__ _Float16 lA[2][BM * LDA];
  __shared__ _Float16 lB[2][BN * LDB];
  int tid = threadIdx.x;
  int wave = tid >> 5, lane = tid & 31;
  int m0 = blockIdx.y * BM;
  int n0 = blockIdx.x * BN;
  int b = n0 >> 12;
  int tt0 = n0 & (TG - 1);

  v8f acc[8] = {};

  auto stage = [&](int buf, int k0) {
    // A tile: 128 rows x 32 halves, 2 async b128 per thread
    int row = tid >> 1;
    int cb = (tid & 1) * 16;
    unsigned lds = (unsigned)(size_t)&lA[buf][row * LDA + cb];
    unsigned long long ga =
        (unsigned long long)(size_t)(A + (m0 + row) * KZ + k0 + cb);
    async_ld_b128(lds, ga);
    async_ld_b128(lds + 16u, ga + 16ull);
    // B tile: gather f32 activations, convert to f16, pack to 2x b128 stores
    int col = tid & 127;
    int kr0 = (tid >> 7) * 16;
    int tt = tt0 + col;
    v8h p0, p1;
    if (k0 < 768) {
      int kk = k0 >> 8;                 // tap index 0..2
      int tsrc = tt + (kk - 1) * dil;   // dilated, zero-padded
      bool ok = (tsrc >= 0) && (tsrc < TG);
      int cbase = (k0 & 255) + kr0;
      const float* xp = x + (b * NCH + cbase) * TG + (ok ? tsrc : 0);
#pragma unroll
      for (int q = 0; q < 8; ++q) {
        p0[q] = ok ? (_Float16)xp[q * TG] : (_Float16)0.f;
        p1[q] = ok ? (_Float16)xp[(q + 8) * TG] : (_Float16)0.f;
      }
    } else {
      int cbase = (k0 - 768) + kr0;
      const float* sp = s + (b * COND + cbase) * TG + tt;
#pragma unroll
      for (int q = 0; q < 8; ++q) {
        p0[q] = (_Float16)sp[q * TG];
        p1[q] = (_Float16)sp[(q + 8) * TG];
      }
    }
    v8h* dst = (v8h*)&lB[buf][col * LDB + kr0];
    dst[0] = p0;
    dst[1] = p1;
  };

  stage(0, 0);
  wait_async0();
  __syncthreads();

  for (int ki = 0; ki < NKZ; ++ki) {
    int cur = ki & 1;
    if (ki + 1 < NKZ) stage(cur ^ 1, (ki + 1) * BK);

    int r = lane & 15;
    int kb = (lane >> 4) * 8;
    v16h af = load_frag(lA[cur], LDA, wave * 16 + r, kb);
#pragma unroll
    for (int j = 0; j < 8; ++j) {
      v16h bf = load_frag(lB[cur], LDB, j * 16 + r, kb);
      acc[j] = __builtin_amdgcn_wmma_f32_16x16x32_f16(
          false, af, false, bf, (short)0, acc[j], false, false);
    }
    wait_async0();
    __syncthreads();
  }

  int mrow = m0 + wave * 16 + (lane >> 4) * 8;
  int ncol = lane & 15;
#pragma unroll
  for (int j = 0; j < 8; ++j) {
    float* zp = z + (b * 512 + mrow) * TG + (tt0 + j * 16 + ncol);
#pragma unroll
    for (int rr = 0; rr < 8; ++rr) zp[rr * TG] = acc[j][rr];
  }
}

// gated activation: acts = tanh(z[:, :256]) * sigmoid(z[:, 256:])
__global__ void gate_kernel(const float* __restrict__ z,
                            float* __restrict__ acts) {
  int idx = blockIdx.x * blockDim.x + threadIdx.x;
  if (idx >= NB * NCH * TG) return;
  int t = idx & (TG - 1);
  int c = (idx >> 12) & 255;
  int b = idx >> 20;
  float zt = z[(b * 512 + c) * TG + t];
  float zs = z[(b * 512 + c + 256) * TG + t];
  acts[idx] = tanhf(zt) * (1.f / (1.f + expf(-zs)));
}

// ---------------------------------------------------------------------------
// res-skip GEMM: rs = rs_w * acts, fused epilogue x += / skip +=
// ---------------------------------------------------------------------------
__global__ __launch_bounds__(256)
void wn_rs_gemm_kernel(const _Float16* __restrict__ A,  // [M, 256] f16
                       const float* __restrict__ acts,  // (NB,256,TG)
                       float* __restrict__ x,           // (NB,256,TG)
                       float* __restrict__ skip,        // (NB,256,TG)
                       int is_last) {
  __shared__ _Float16 lA[2][BM * LDA];
  __shared__ _Float16 lB[2][BN * LDB];
  int tid = threadIdx.x;
  int wave = tid >> 5, lane = tid & 31;
  int m0 = blockIdx.y * BM;
  int n0 = blockIdx.x * BN;
  int b = n0 >> 12;
  int tt0 = n0 & (TG - 1);

  v8f acc[8] = {};

  auto stage = [&](int buf, int k0) {
    int row = tid >> 1;
    int cb = (tid & 1) * 16;
    unsigned lds = (unsigned)(size_t)&lA[buf][row * LDA + cb];
    unsigned long long ga =
        (unsigned long long)(size_t)(A + (m0 + row) * NCH + k0 + cb);
    async_ld_b128(lds, ga);
    async_ld_b128(lds + 16u, ga + 16ull);
    int col = tid & 127;
    int kr0 = (tid >> 7) * 16;
    int tt = tt0 + col;
    const float* sp = acts + (b * NCH + k0 + kr0) * TG + tt;
    v8h p0, p1;
#pragma unroll
    for (int q = 0; q < 8; ++q) {
      p0[q] = (_Float16)sp[q * TG];
      p1[q] = (_Float16)sp[(q + 8) * TG];
    }
    v8h* dst = (v8h*)&lB[buf][col * LDB + kr0];
    dst[0] = p0;
    dst[1] = p1;
  };

  stage(0, 0);
  wait_async0();
  __syncthreads();

  for (int ki = 0; ki < NKR; ++ki) {
    int cur = ki & 1;
    if (ki + 1 < NKR) stage(cur ^ 1, (ki + 1) * BK);

    int r = lane & 15;
    int kb = (lane >> 4) * 8;
    v16h af = load_frag(lA[cur], LDA, wave * 16 + r, kb);
#pragma unroll
    for (int j = 0; j < 8; ++j) {
      v16h bf = load_frag(lB[cur], LDB, j * 16 + r, kb);
      acc[j] = __builtin_amdgcn_wmma_f32_16x16x32_f16(
          false, af, false, bf, (short)0, acc[j], false, false);
    }
    wait_async0();
    __syncthreads();
  }

  int mrow = m0 + wave * 16 + (lane >> 4) * 8;
  int ncol = lane & 15;
#pragma unroll
  for (int j = 0; j < 8; ++j) {
    int t = tt0 + j * 16 + ncol;
#pragma unroll
    for (int rr = 0; rr < 8; ++rr) {
      int co = mrow + rr;
      float val = acc[j][rr];
      if (is_last) {
        skip[(b * NCH + co) * TG + t] += val;
      } else if (co < NCH) {
        x[(b * NCH + co) * TG + t] += val;
      } else {
        skip[(b * NCH + co - NCH) * TG + t] += val;
      }
    }
  }
}

// end conv: o[b,j,t] = sum_c end_w[j*256+c] * skip[b,c,t]   (j < 2*nh <= 8)
__global__ void end_conv_kernel(const float* __restrict__ w,
                                const float* __restrict__ skip,
                                float* __restrict__ o, int nh2) {
  int idx = blockIdx.x * blockDim.x + threadIdx.x;
  if (idx >= NB * nh2 * TG) return;
  int t = idx & (TG - 1);
  int j = (idx >> 12) % nh2;
  int b = idx / (nh2 * TG);
  float acc = 0.f;
  for (int c = 0; c < NCH; ++c) acc += w[j * NCH + c] * skip[(b * NCH + c) * TG + t];
  o[(b * 8 + j) * TG + t] = acc;
}

// affine coupling: a1 = exp(log_s)*a1 + b; emit log_s to output
__global__ void affine_kernel(const float* __restrict__ o,
                              float* __restrict__ a,
                              float* __restrict__ out_ls, int nh) {
  int idx = blockIdx.x * blockDim.x + threadIdx.x;
  if (idx >= NB * nh * TG) return;
  int t = idx & (TG - 1);
  int h = (idx >> 12) % nh;
  int b = idx / (nh * TG);
  float ls = o[(b * 8 + nh + h) * TG + t];
  float bb = o[(b * 8 + h) * TG + t];
  float* ap = &a[(b * 8 + nh + h) * TG + t];
  *ap = expf(ls) * (*ap) + bb;
  out_ls[(b * nh + h) * TG + t] = ls;
}

// copy nch channels of a (8-ch layout) to dst (8-ch layout)
__global__ void copy_ch_kernel(const float* __restrict__ a,
                               float* __restrict__ dst,
                               int nch, int src_ch0, int dst_ch0) {
  int idx = blockIdx.x * blockDim.x + threadIdx.x;
  if (idx >= NB * nch * TG) return;
  int t = idx & (TG - 1);
  int c = (idx >> 12) % nch;
  int b = idx / (nch * TG);
  dst[(b * 8 + dst_ch0 + c) * TG + t] = a[(b * 8 + src_ch0 + c) * TG + t];
}

// ---------------------------------------------------------------------------
// Host orchestration
// ---------------------------------------------------------------------------
static inline int cdiv(int a, int b) { return (a + b - 1) / b; }

extern "C" void kernel_launch(void* const* d_in, const int* in_sizes, int n_in,
                              void* d_out, int out_size, void* d_ws, size_t ws_size,
                              hipStream_t stream) {
  (void)in_sizes; (void)n_in; (void)out_size; (void)ws_size;
  const float* spect = (const float*)d_in[0];
  const float* audio = (const float*)d_in[1];
  // params pytree (sorted keys): flows[12]{W, wn{end_b,end_w,layers[8]{cond_b,
  // cond_w,in_b,in_w,rs_b,rs_w}, start_b, start_w}}, upsample_b, upsample_w
  auto F = [&](int i) { return (const float*)d_in[i]; };
  const float* ups_w = F(639);

  // workspace layout (f32 then f16)
  float* ws = (float*)d_ws;
  float* s    = ws;                        // 2*640*4096
  float* x    = s    + NB * COND * TG;     // 2*256*4096
  float* z    = x    + NB * NCH * TG;      // 2*512*4096
  float* acts = z    + NB * 512 * TG;      // 2*256*4096
  float* skip = acts + NB * NCH * TG;      // 2*256*4096
  float* oend = skip + NB * NCH * TG;      // 2*8*4096
  float* aA   = oend + NB * 8 * TG;        // 2*8*4096
  float* aB   = aA   + NB * 8 * TG;        // 2*8*4096
  _Float16* wA  = (_Float16*)(aB + NB * 8 * TG);  // 512*1408
  _Float16* wRs = wA + 512 * KZ;                  // 512*256

  float* out_z  = (float*)d_out;                 // (NB, 8, TG)
  float* out_ls = out_z + NB * 8 * TG;           // 12 flows of (NB, nh, TG)
  float* out_ld = out_ls + NB * 36 * TG;         // 12 scalars

  const int T256 = 256;

  upsample_cond_kernel<<<cdiv(NB * COND * TG, T256), T256, 0, stream>>>(spect, ups_w, s);
  reshape_audio_kernel<<<cdiv(NB * 8 * TG, T256), T256, 0, stream>>>(audio, aA);

  float* acur = aA;
  float* anxt = aB;
  int nrem = 8;
  int early_dst = 0;
  int ls_off = 0;

  for (int kf = 0; kf < 12; ++kf) {
    int src_ch0 = 0;
    if (kf % 4 == 0 && kf > 0) {
      copy_ch_kernel<<<cdiv(NB * 2 * TG, T256), T256, 0, stream>>>(
          acur, out_z, 2, 0, early_dst);
      early_dst += 2;
      src_ch0 = 2;
      nrem -= 2;
    }
    int nh = nrem / 2;
    int p = 2 + kf * 53;
    const float* W = F(p);

    logdet_kernel<<<1, 32, 0, stream>>>(W, out_ld + kf, nrem);
    invconv_kernel<<<cdiv(NB * TG, T256), T256, 0, stream>>>(
        W, acur, anxt, nrem, src_ch0);
    { float* t = acur; acur = anxt; anxt = t; }

    start_conv_kernel<<<cdiv(NB * NCH * TG, T256), T256, 0, stream>>>(
        F(p + 52), acur, x, nh);
    zero_kernel<<<cdiv(NB * NCH * TG, T256), T256, 0, stream>>>(skip, NB * NCH * TG);

    for (int i = 0; i < 8; ++i) {
      int q = p + 3 + 6 * i;
      pack_wz_kernel<<<cdiv(512 * KZ, T256), T256, 0, stream>>>(F(q + 3), F(q + 1), wA);
      wn_z_gemm_kernel<<<dim3(NB * TG / BN, 512 / BM), T256, 0, stream>>>(
          wA, x, s, z, 1 << i);
      gate_kernel<<<cdiv(NB * NCH * TG, T256), T256, 0, stream>>>(z, acts);
      int rsco = (i < 7) ? 512 : 256;
      pack_f16_kernel<<<cdiv(rsco * NCH, T256), T256, 0, stream>>>(
          F(q + 5), wRs, rsco * NCH);
      wn_rs_gemm_kernel<<<dim3(NB * TG / BN, rsco / BM), T256, 0, stream>>>(
          wRs, acts, x, skip, (i == 7) ? 1 : 0);
    }

    end_conv_kernel<<<cdiv(NB * 2 * nh * TG, T256), T256, 0, stream>>>(
        F(p + 2), skip, oend, 2 * nh);
    affine_kernel<<<cdiv(NB * nh * TG, T256), T256, 0, stream>>>(
        oend, acur, out_ls + ls_off, nh);
    ls_off += NB * nh * TG;
  }

  copy_ch_kernel<<<cdiv(NB * 4 * TG, T256), T256, 0, stream>>>(acur, out_z, 4, 0, 4);
}
